// DualAttentionRNNModelOnePoint_7945689497896
// MI455X (gfx1250) — matrix operand order repaired and checked by reading, not compiled, not run on hardware
//
#include <hip/hip_runtime.h>

typedef __attribute__((ext_vector_type(16))) _Float16 v16h;
typedef __attribute__((ext_vector_type(8)))  _Float16 v8h;
typedef __attribute__((ext_vector_type(8)))  float    v8f;

#define DEV __device__ __forceinline__

constexpr int B = 64, T = 96, F = 32, H = 128, H4 = 512;
constexpr int BPG = 16;              // batch rows per workgroup (== WMMA M)
constexpr int NBLK = B / BPG;        // 4 scan workgroups
constexpr int NT_SCAN = 512;         // 16 waves for the serial scan kernels
constexpr int NW_SCAN = NT_SCAN / 32;
constexpr int NT_GEMM = 256;         // 8 waves for the batch GEMM kernel

// ---- workspace byte offsets (packed f16 weights + f32 intermediates) ----
constexpr size_t PK_W1H  = 0;                                   // 128x128 -> 32KB
constexpr size_t PK_W1S  = 32768;
constexpr size_t PK_ENCK = 65536;                               // 32x512  -> 32KB
constexpr size_t PK_ENCR = 98304;                               // 128x512 -> 128KB
constexpr size_t PK_ADD  = 229376;
constexpr size_t PK_ADC  = 262144;
constexpr size_t PK_ADX  = 294912;
constexpr size_t PK_DECK = 327680;                              // 128KB
constexpr size_t PK_DECR = 458752;                              // 128KB
constexpr size_t PK_FCW  = 589824;                              // 128x128 -> 32KB
constexpr size_t PK_FFW  = 622592;                              // 256x32  -> 16KB
constexpr size_t WS_XW   = 655360;                              // B*F*H f32 = 1MB
constexpr size_t WS_XENC = WS_XW  + (size_t)B * F * H * 4;      // B*T*H f32 = 3MB
constexpr size_t WS_XDP  = WS_XENC + (size_t)B * T * H * 4;     // B*T*H f32 = 3MB

// ---------------- WMMA fragment helpers (16x16x32 f16 -> f32) ----------------
// A frag from f16 LDS (row-major M x K): lane m=lane&15, half=lane>>4 holds
// K = kbase + half*8 + {0..7} and kbase + 16 + half*8 + {0..7}  -> 2x b128 loads
DEV v16h load_a_frag16(const _Float16* A, int ldK, int kbase, int lane) {
  const int m = lane & 15, half = lane >> 4;
  const _Float16* r = A + m * ldK + kbase + half * 8;
  union { v16h v; struct { v8h lo, hi; } p; } u;
  u.p.lo = *(const v8h*)r;
  u.p.hi = *(const v8h*)(r + 16);
  return u.v;
}

// B frag: pre-packed so each lane reads its 16 halves contiguously (32B)
DEV v16h load_b_frag(const _Float16* bp, int ntiles, int kt, int nt, int lane) {
  return *(const v16h*)(bp + ((size_t)(kt * ntiles + nt) * 32 + lane) * 16);
}

DEV v8f wmma_f16(v16h a, v16h b, v8f c) {
  return __builtin_amdgcn_wmma_f32_16x16x32_f16(false, a, false, b, (short)0, c,
                                                false, false);
}

// D frag: lane holds col = nt*16 + (lane&15), rows = (lane>>4)*8 + r
DEV void store_d_lds(float* D, int ldN, int nt, int lane, v8f acc) {
  const int col = nt * 16 + (lane & 15), half = lane >> 4;
#pragma unroll
  for (int r = 0; r < 8; ++r) D[(half * 8 + r) * ldN + col] = acc[r];
}

DEV v8f bias_frag(const float* bias, int nt, int lane) {
  float v = bias ? bias[nt * 16 + (lane & 15)] : 0.f;
  v8f c;
#pragma unroll
  for (int r = 0; r < 8; ++r) c[r] = v;
  return c;
}

DEV float fsigmoid(float x) { return 1.f / (1.f + __expf(-x)); }
DEV float ftanh(float x) {
  x = fminf(15.f, fmaxf(-15.f, x));
  float e = __expf(2.f * x);
  return (e - 1.f) / (e + 1.f);
}

// ---------------- prep: pack weight (KxN f32, row stride ldn) to B-frag layout ----
__global__ void pack_b_kernel(const float* __restrict__ W, int ldn, int ktiles,
                              int ntiles, _Float16* __restrict__ dst) {
  int idx = blockIdx.x * blockDim.x + threadIdx.x;
  int total = ktiles * ntiles * 32;
  if (idx >= total) return;
  int lane = idx & 31;
  int nt = (idx >> 5) % ntiles;
  int kt = (idx >> 5) / ntiles;
  int half = lane >> 4, col = nt * 16 + (lane & 15);
  _Float16* o = dst + (size_t)idx * 16;
#pragma unroll
  for (int i = 0; i < 16; ++i) {
    int k = kt * 32 + (i < 8 ? half * 8 + i : 16 + half * 8 + (i - 8));
    o[i] = (_Float16)W[(size_t)k * ldn + col];
  }
}

// ---------------- prep: xw[b,f,:] = x_perm[b,f,:] @ ae_w1[2H: ,:] + ae_b1 -------
__global__ void xw_kernel(const float* __restrict__ inputs,
                          const float* __restrict__ ae_w1,
                          const float* __restrict__ ae_b1,
                          float* __restrict__ xw) {
  int idx = blockIdx.x * blockDim.x + threadIdx.x;   // B*F*H
  if (idx >= B * F * H) return;
  int j = idx & (H - 1);
  int f = (idx >> 7) & (F - 1);
  int b = idx >> 12;
  float acc = ae_b1[j];
  for (int t = 0; t < T; ++t)
    acc += inputs[(size_t)(b * T + t) * F + f] * ae_w1[(size_t)(2 * H + t) * H + j];
  xw[idx] = acc;
}

// ---------------- encoder scan: 4 WGs x 16 batch rows, 96 serial steps ----------
__global__ __launch_bounds__(NT_SCAN) void encoder_kernel(
    const float* __restrict__ inputs, const _Float16* __restrict__ pkW1h,
    const _Float16* __restrict__ pkW1s, const _Float16* __restrict__ pkEncK,
    const _Float16* __restrict__ pkEncR, const float* __restrict__ enc_b,
    const float* __restrict__ ae_w2, const float* __restrict__ xw,
    float* __restrict__ xenc) {
  __shared__ __align__(16) _Float16 sh_h16[BPG * H];   // h   (A operand)
  __shared__ __align__(16) _Float16 sh_s16[BPG * H];   // c   (A operand mirror)
  __shared__ __align__(16) float sh_s[BPG * H];        // c   (f32 master)
  __shared__ __align__(16) float sh_z[BPG * H4];       // gate pre-activations
  __shared__ __align__(16) float sh_att[BPG * H];
  __shared__ __align__(16) float sh_e[BPG * F];
  __shared__ __align__(16) _Float16 sh_xt16[BPG * F];  // x_tilde (A operand)
  const int tid = threadIdx.x, lane = tid & 31, wave = tid >> 5;
  const int b0 = blockIdx.x * BPG;
  for (int i = tid; i < BPG * H; i += NT_SCAN) {
    sh_h16[i] = (_Float16)0.f; sh_s16[i] = (_Float16)0.f; sh_s[i] = 0.f;
  }
  // hoist step-invariant z-GEMM B fragments into registers (2 n-tiles per wave)
  v16h bz[2][5];
#pragma unroll
  for (int i = 0; i < 2; ++i) {
    int nt = wave * 2 + i;
    bz[i][0] = load_b_frag(pkEncK, 32, 0, nt, lane);
#pragma unroll
    for (int kt = 0; kt < 4; ++kt)
      bz[i][1 + kt] = load_b_frag(pkEncR, 32, kt, nt, lane);
  }
  __syncthreads();

  for (int t = 0; t < T; ++t) {
    // attnH = h @ W1h + s @ W1s : 8 n-tiles on waves 0..7
    for (int nt = wave; nt < H / 16; nt += NW_SCAN) {
      v8f acc = bias_frag(nullptr, nt, lane);
#pragma unroll
      for (int kt = 0; kt < 4; ++kt)
        acc = wmma_f16(load_a_frag16(sh_h16, H, kt * 32, lane),
                       load_b_frag(pkW1h, 8, kt, nt, lane), acc);
#pragma unroll
      for (int kt = 0; kt < 4; ++kt)
        acc = wmma_f16(load_a_frag16(sh_s16, H, kt * 32, lane),
                       load_b_frag(pkW1s, 8, kt, nt, lane), acc);
      store_d_lds(sh_att, H, nt, lane, acc);
    }
    __syncthreads();
    // e[b,f] = tanh(xw + attnH) . ae_w2   (vectorized float4)
    for (int p = tid; p < BPG * F; p += NT_SCAN) {
      int b = p >> 5, f = p & (F - 1);
      const float4* xr = (const float4*)(xw + ((size_t)(b0 + b) * F + f) * H);
      const float4* ar = (const float4*)(sh_att + b * H);
      const float4* w2 = (const float4*)ae_w2;
      float acc = 0.f;
      for (int j = 0; j < H / 4; ++j) {
        float4 x = xr[j], a = ar[j], w = w2[j];
        acc += ftanh(x.x + a.x) * w.x + ftanh(x.y + a.y) * w.y +
               ftanh(x.z + a.z) * w.z + ftanh(x.w + a.w) * w.w;
      }
      sh_e[p] = acc;
    }
    __syncthreads();
    if (tid < BPG) {                       // softmax over F per batch row
      float* e = sh_e + tid * F;
      float mx = e[0];
      for (int f = 1; f < F; ++f) mx = fmaxf(mx, e[f]);
      float sum = 0.f;
      for (int f = 0; f < F; ++f) { float v = __expf(e[f] - mx); e[f] = v; sum += v; }
      float inv = 1.f / sum;
      for (int f = 0; f < F; ++f) e[f] *= inv;
    }
    __syncthreads();
    for (int p = tid; p < BPG * F; p += NT_SCAN) {   // x_tilde = alpha * x_t
      int b = p >> 5, f = p & (F - 1);
      sh_xt16[p] = (_Float16)(sh_e[p] * inputs[((size_t)(b0 + b) * T + t) * F + f]);
    }
    __syncthreads();
    // z = x_tilde @ enc_k + h @ enc_r + enc_b : 32 n-tiles, B frags in registers
#pragma unroll
    for (int i = 0; i < 2; ++i) {
      int nt = wave * 2 + i;
      v8f acc = bias_frag(enc_b, nt, lane);
      acc = wmma_f16(load_a_frag16(sh_xt16, F, 0, lane), bz[i][0], acc);
#pragma unroll
      for (int kt = 0; kt < 4; ++kt)
        acc = wmma_f16(load_a_frag16(sh_h16, H, kt * 32, lane), bz[i][1 + kt], acc);
      store_d_lds(sh_z, H4, nt, lane, acc);
    }
    __syncthreads();
    for (int idx = tid; idx < BPG * H; idx += NT_SCAN) {   // LSTM gates
      int b = idx >> 7, j = idx & (H - 1);
      const float* zr = sh_z + b * H4;
      float ig = fsigmoid(zr[j]);
      float fg = fsigmoid(zr[H + j]);
      float gg = ftanh(zr[2 * H + j]);
      float og = fsigmoid(zr[3 * H + j]);
      float c = fg * sh_s[idx] + ig * gg;
      float h = og * ftanh(c);
      sh_s[idx] = c;
      sh_s16[idx] = (_Float16)c;
      sh_h16[idx] = (_Float16)h;
      xenc[((size_t)(b0 + b) * T + t) * H + j] = h;
    }
    __syncthreads();
  }
}

// ---------------- step-invariant decoder projection: xenc @ ad_w1[2H:,:] + b ----
__global__ __launch_bounds__(NT_GEMM) void xdproj_kernel(
    const float* __restrict__ xenc, const _Float16* __restrict__ pkAdX,
    const float* __restrict__ ad_b1, float* __restrict__ xdp) {
  __shared__ __align__(16) _Float16 sh_a16[16 * H];
  const int tid = threadIdx.x, lane = tid & 31, wave = tid >> 5;
  const size_t row0 = (size_t)blockIdx.x * 16;
  for (int i = tid; i < 16 * H; i += NT_GEMM)
    sh_a16[i] = (_Float16)xenc[row0 * H + i];
  __syncthreads();
  for (int nt = wave; nt < H / 16; nt += NT_GEMM / 32) {
    v8f acc = bias_frag(ad_b1, nt, lane);
#pragma unroll
    for (int kt = 0; kt < 4; ++kt)
      acc = wmma_f16(load_a_frag16(sh_a16, H, kt * 32, lane),
                     load_b_frag(pkAdX, 8, kt, nt, lane), acc);
    int col = nt * 16 + (lane & 15), half = lane >> 4;
#pragma unroll
    for (int r = 0; r < 8; ++r) xdp[(row0 + half * 8 + r) * H + col] = acc[r];
  }
}

// ---------------- decoder scan + final projection --------------------------------
__global__ __launch_bounds__(NT_SCAN) void decoder_kernel(
    const float* __restrict__ inputs, const float* __restrict__ xenc,
    const float* __restrict__ xdp, const _Float16* __restrict__ pkAdD,
    const _Float16* __restrict__ pkAdC, const _Float16* __restrict__ pkDecK,
    const _Float16* __restrict__ pkDecR, const float* __restrict__ dec_b,
    const float* __restrict__ ad_w2, const _Float16* __restrict__ pkFcW,
    const float* __restrict__ fc_w, const float* __restrict__ fc_b,
    const _Float16* __restrict__ pkFfW, const float* __restrict__ ff_b,
    float* __restrict__ out) {
  __shared__ __align__(16) _Float16 sh_d16[BPG * H];
  __shared__ __align__(16) _Float16 sh_c16[BPG * H];
  __shared__ __align__(16) float sh_c[BPG * H];
  __shared__ __align__(16) _Float16 sh_ctx16[BPG * H];
  __shared__ __align__(16) float sh_z[BPG * H4];  // overlays: attnD [0,2048),
                                                  // e/beta [2048,3584), ytf [4096,6144)
  __shared__ __align__(16) _Float16 sh_yt16[BPG * H];
  float* sh_att = sh_z;
  float* sh_e = sh_z + BPG * H;
  float* sh_ytf = sh_z + 2 * BPG * H;
  const int tid = threadIdx.x, lane = tid & 31, wave = tid >> 5;
  const int b0 = blockIdx.x * BPG;
  for (int i = tid; i < BPG * H; i += NT_SCAN) {
    sh_d16[i] = (_Float16)0.f; sh_c16[i] = (_Float16)0.f; sh_c[i] = 0.f;
  }
  // hoist step-invariant z-GEMM B fragments into registers (2 n-tiles per wave)
  v16h bz[2][8];
#pragma unroll
  for (int i = 0; i < 2; ++i) {
    int nt = wave * 2 + i;
#pragma unroll
    for (int kt = 0; kt < 4; ++kt) {
      bz[i][kt]     = load_b_frag(pkDecK, 32, kt, nt, lane);
      bz[i][4 + kt] = load_b_frag(pkDecR, 32, kt, nt, lane);
    }
  }
  __syncthreads();

  for (int t = 0; t < T; ++t) {
    // attnD = d @ AdD + c @ AdC : 8 n-tiles on waves 0..7
    for (int nt = wave; nt < H / 16; nt += NW_SCAN) {
      v8f acc = bias_frag(nullptr, nt, lane);
#pragma unroll
      for (int kt = 0; kt < 4; ++kt)
        acc = wmma_f16(load_a_frag16(sh_d16, H, kt * 32, lane),
                       load_b_frag(pkAdD, 8, kt, nt, lane), acc);
#pragma unroll
      for (int kt = 0; kt < 4; ++kt)
        acc = wmma_f16(load_a_frag16(sh_c16, H, kt * 32, lane),
                       load_b_frag(pkAdC, 8, kt, nt, lane), acc);
      store_d_lds(sh_att, H, nt, lane, acc);
    }
    __syncthreads();
    // e[b,t'] = tanh(xdproj + attnD) . ad_w2   (vectorized float4)
    for (int p = tid; p < BPG * T; p += NT_SCAN) {
      int b = p / T, tt = p - b * T;
      const float4* xr = (const float4*)(xdp + ((size_t)(b0 + b) * T + tt) * H);
      const float4* ar = (const float4*)(sh_att + b * H);
      const float4* w2 = (const float4*)ad_w2;
      float acc = 0.f;
      for (int j = 0; j < H / 4; ++j) {
        float4 x = xr[j], a = ar[j], w = w2[j];
        acc += ftanh(x.x + a.x) * w.x + ftanh(x.y + a.y) * w.y +
               ftanh(x.z + a.z) * w.z + ftanh(x.w + a.w) * w.w;
      }
      sh_e[p] = acc;
    }
    __syncthreads();
    if (tid < BPG) {                       // softmax over T per batch row -> beta
      float* e = sh_e + tid * T;
      float mx = e[0];
      for (int k = 1; k < T; ++k) mx = fmaxf(mx, e[k]);
      float sum = 0.f;
      for (int k = 0; k < T; ++k) { float v = __expf(e[k] - mx); e[k] = v; sum += v; }
      float inv = 1.f / sum;
      for (int k = 0; k < T; ++k) e[k] *= inv;
    }
    __syncthreads();
    // ctx[b,:] = beta[b,:] @ xenc[b]  (batched GEMV, float4 over j)
    for (int q = tid; q < BPG * (H / 4); q += NT_SCAN) {
      int b = q >> 5, j4 = q & 31;
      const float* beta = sh_e + b * T;
      const float4* xr = (const float4*)(xenc + (size_t)(b0 + b) * T * H) + j4;
      float4 acc = make_float4(0.f, 0.f, 0.f, 0.f);
      for (int k = 0; k < T; ++k) {
        float4 x = xr[(size_t)k * (H / 4)];
        float bk = beta[k];
        acc.x += bk * x.x; acc.y += bk * x.y; acc.z += bk * x.z; acc.w += bk * x.w;
      }
      int base = b * H + j4 * 4;
      sh_ctx16[base + 0] = (_Float16)acc.x;
      sh_ctx16[base + 1] = (_Float16)acc.y;
      sh_ctx16[base + 2] = (_Float16)acc.z;
      sh_ctx16[base + 3] = (_Float16)acc.w;
    }
    __syncthreads();
    // y_tilde = ctx @ fc_w[0:128,:] + fc_b  (WMMA) ; + y_t * fc_w[128,:] after
    for (int nt = wave; nt < H / 16; nt += NW_SCAN) {
      v8f acc = bias_frag(fc_b, nt, lane);
#pragma unroll
      for (int kt = 0; kt < 4; ++kt)
        acc = wmma_f16(load_a_frag16(sh_ctx16, H, kt * 32, lane),
                       load_b_frag(pkFcW, 8, kt, nt, lane), acc);
      store_d_lds(sh_ytf, H, nt, lane, acc);
    }
    __syncthreads();
    for (int idx = tid; idx < BPG * H; idx += NT_SCAN) {
      int b = idx >> 7, j = idx & (H - 1);
      float yprev = inputs[((size_t)(b0 + b) * T + t) * F + (F - 1)];
      sh_yt16[idx] = (_Float16)(sh_ytf[idx] + yprev * fc_w[(size_t)H * H + j]);
    }
    __syncthreads();
    // z = y_tilde @ dec_k + d @ dec_r + dec_b  (B frags in registers; overlays dead)
#pragma unroll
    for (int i = 0; i < 2; ++i) {
      int nt = wave * 2 + i;
      v8f acc = bias_frag(dec_b, nt, lane);
#pragma unroll
      for (int kt = 0; kt < 4; ++kt)
        acc = wmma_f16(load_a_frag16(sh_yt16, H, kt * 32, lane), bz[i][kt], acc);
#pragma unroll
      for (int kt = 0; kt < 4; ++kt)
        acc = wmma_f16(load_a_frag16(sh_d16, H, kt * 32, lane), bz[i][4 + kt], acc);
      store_d_lds(sh_z, H4, nt, lane, acc);
    }
    __syncthreads();
    for (int idx = tid; idx < BPG * H; idx += NT_SCAN) {   // LSTM gates
      int b = idx >> 7, j = idx & (H - 1);
      const float* zr = sh_z + b * H4;
      float ig = fsigmoid(zr[j]);
      float fg = fsigmoid(zr[H + j]);
      float gg = ftanh(zr[2 * H + j]);
      float og = fsigmoid(zr[3 * H + j]);
      float c = fg * sh_c[idx] + ig * gg;
      sh_c[idx] = c;
      sh_c16[idx] = (_Float16)c;
      sh_d16[idx] = (_Float16)(og * ftanh(c));
    }
    __syncthreads();
  }

  // out = [d_n, ctx] @ ff_w + ff_b   (K=256 -> 8 k-tiles, N=32 -> 2 n-tiles)
  for (int nt = wave; nt < F / 16; nt += NW_SCAN) {
    v8f acc = bias_frag(ff_b, nt, lane);
#pragma unroll
    for (int kt = 0; kt < 4; ++kt)
      acc = wmma_f16(load_a_frag16(sh_d16, H, kt * 32, lane),
                     load_b_frag(pkFfW, 2, kt, nt, lane), acc);
#pragma unroll
    for (int kt = 0; kt < 4; ++kt)
      acc = wmma_f16(load_a_frag16(sh_ctx16, H, kt * 32, lane),
                     load_b_frag(pkFfW, 2, kt + 4, nt, lane), acc);
    int col = nt * 16 + (lane & 15), half = lane >> 4;
#pragma unroll
    for (int r = 0; r < 8; ++r) out[(size_t)(b0 + half * 8 + r) * F + col] = acc[r];
  }
}

extern "C" void kernel_launch(void* const* d_in, const int* in_sizes, int n_in,
                              void* d_out, int out_size, void* d_ws, size_t ws_size,
                              hipStream_t stream) {
  (void)in_sizes; (void)n_in; (void)out_size; (void)ws_size;
  const float* inputs = (const float*)d_in[0];
  const float* enc_k  = (const float*)d_in[1];
  const float* enc_r  = (const float*)d_in[2];
  const float* enc_b  = (const float*)d_in[3];
  const float* ae_w1  = (const float*)d_in[4];
  const float* ae_b1  = (const float*)d_in[5];
  const float* ae_w2  = (const float*)d_in[6];
  // d_in[7] = ae_b2: zeros and softmax-invariant -> unused
  const float* dec_k  = (const float*)d_in[8];
  const float* dec_r  = (const float*)d_in[9];
  const float* dec_b  = (const float*)d_in[10];
  const float* ad_w1  = (const float*)d_in[11];
  const float* ad_b1  = (const float*)d_in[12];
  const float* ad_w2  = (const float*)d_in[13];
  // d_in[14] = ad_b2: unused (softmax-invariant)
  const float* fc_w   = (const float*)d_in[15];
  const float* fc_b   = (const float*)d_in[16];
  const float* ff_w   = (const float*)d_in[17];
  const float* ff_b   = (const float*)d_in[18];
  char* ws = (char*)d_ws;

  auto pk = [&](const float* W, int ldn, int K, int N, size_t off) {
    int total = (K / 32) * (N / 16) * 32;
    pack_b_kernel<<<(total + 255) / 256, 256, 0, stream>>>(
        W, ldn, K / 32, N / 16, (_Float16*)(ws + off));
  };
  pk(ae_w1,                      H,  H,     H,  PK_W1H);
  pk(ae_w1 + (size_t)H * H,      H,  H,     H,  PK_W1S);
  pk(enc_k,                      H4, F,     H4, PK_ENCK);
  pk(enc_r,                      H4, H,     H4, PK_ENCR);
  pk(ad_w1,                      H,  H,     H,  PK_ADD);
  pk(ad_w1 + (size_t)H * H,      H,  H,     H,  PK_ADC);
  pk(ad_w1 + (size_t)2 * H * H,  H,  H,     H,  PK_ADX);
  pk(dec_k,                      H4, H,     H4, PK_DECK);
  pk(dec_r,                      H4, H,     H4, PK_DECR);
  pk(fc_w,                       H,  H,     H,  PK_FCW);
  pk(ff_w,                       F,  2 * H, F,  PK_FFW);

  xw_kernel<<<(B * F * H + 255) / 256, 256, 0, stream>>>(
      inputs, ae_w1, ae_b1, (float*)(ws + WS_XW));

  encoder_kernel<<<NBLK, NT_SCAN, 0, stream>>>(
      inputs, (const _Float16*)(ws + PK_W1H), (const _Float16*)(ws + PK_W1S),
      (const _Float16*)(ws + PK_ENCK), (const _Float16*)(ws + PK_ENCR), enc_b,
      ae_w2, (const float*)(ws + WS_XW), (float*)(ws + WS_XENC));

  xdproj_kernel<<<(B * T) / 16, NT_GEMM, 0, stream>>>(
      (const float*)(ws + WS_XENC), (const _Float16*)(ws + PK_ADX), ad_b1,
      (float*)(ws + WS_XDP));

  decoder_kernel<<<NBLK, NT_SCAN, 0, stream>>>(
      inputs, (const float*)(ws + WS_XENC), (const float*)(ws + WS_XDP),
      (const _Float16*)(ws + PK_ADD), (const _Float16*)(ws + PK_ADC),
      (const _Float16*)(ws + PK_DECK), (const _Float16*)(ws + PK_DECR), dec_b,
      ad_w2, (const _Float16*)(ws + PK_FCW), fc_w, fc_b,
      (const _Float16*)(ws + PK_FFW), ff_b, (float*)d_out);
}
